// MambaBlock_27943057228439
// MI455X (gfx1250) — matrix operand order repaired and checked by reading, not compiled, not run on hardware
//
#include <hip/hip_runtime.h>
#include <hip/hip_bf16.h>
#include <math.h>

// ---------------------------------------------------------------------------
// Mamba block forward for gfx1250 (MI455X).
// BF16 WMMA GEMMs with LDS double-buffered tiles + software-pipelined scan.
// ---------------------------------------------------------------------------

typedef __bf16 bf16;
typedef __attribute__((ext_vector_type(16))) __bf16 v16bf;
typedef __attribute__((ext_vector_type(8)))  __bf16 v8bf;
typedef __attribute__((ext_vector_type(8)))  float  v8f;

#define D_MODEL 1024
#define D_INNER 2048
#define D_STATE 16
#define D_CONV  4
#define BATCH   2
#define SEQ     1024
#define M_ROWS  (BATCH * SEQ)          // 2048
#define N_IN    (2 * D_INNER)          // 4096
#define N_X     (D_INNER + 2*D_STATE)  // 2080
#define N_X_PAD 2176                   // padded to multiple of 128
#define LDSP    40                     // 32 K-elements + 8 pad (80B row stride)

static __device__ __forceinline__ float siluf(float x) {
    return x / (1.f + expf(-x));
}
static __device__ __forceinline__ float softplusf(float x) {
    return x > 20.f ? x : log1pf(expf(x));
}
static __device__ __forceinline__ v16bf combine16(v8bf lo, v8bf hi) {
    return __builtin_shufflevector(lo, hi, 0,1,2,3,4,5,6,7,8,9,10,11,12,13,14,15);
}

// ---------------------------------------------------------------------------
// Weight convert + transpose via 32x32 LDS tile (coalesced both directions):
// W[K,N] f32 row-major -> WT[Npad,K] bf16, zero padded for n >= N.
// grid = (K/32, Npad/32), block = 256.
// ---------------------------------------------------------------------------
__global__ __launch_bounds__(256)
void wt_convert_kernel(const float* __restrict__ W, bf16* __restrict__ WT,
                       int K, int N, int Npad) {
    __shared__ float tile[32][33];
    const int tx = threadIdx.x & 31;
    const int ty = threadIdx.x >> 5;   // 0..7
    const int k0 = blockIdx.x * 32;
    const int n0 = blockIdx.y * 32;
    #pragma unroll
    for (int i = 0; i < 32; i += 8) {
        const int k = k0 + ty + i;
        const int n = n0 + tx;
        tile[ty + i][tx] = (n < N) ? W[(size_t)k * N + n] : 0.f;
    }
    __syncthreads();
    #pragma unroll
    for (int i = 0; i < 32; i += 8) {
        const int n = n0 + ty + i;
        const int k = k0 + tx;
        WT[(size_t)n * K + k] = (bf16)tile[tx][ty + i];
    }
}

// ---------------------------------------------------------------------------
// RMSNorm: one block per row of x [M_ROWS, D_MODEL]; writes bf16.
// ---------------------------------------------------------------------------
__global__ __launch_bounds__(256)
void rmsnorm_kernel(const float* __restrict__ x, const float* __restrict__ w,
                    bf16* __restrict__ xn) {
    __shared__ float red[256];
    const int row = blockIdx.x;
    const int tid = threadIdx.x;
    const float* xr = x + (size_t)row * D_MODEL;
    float s = 0.f;
    float v[4];
    #pragma unroll
    for (int i = 0; i < 4; ++i) {
        v[i] = xr[tid + i * 256];
        s += v[i] * v[i];
    }
    red[tid] = s;
    __syncthreads();
    for (int off = 128; off > 0; off >>= 1) {
        if (tid < off) red[tid] += red[tid + off];
        __syncthreads();
    }
    const float scale = rsqrtf(red[0] * (1.f / D_MODEL) + 1e-5f);
    bf16* o = xn + (size_t)row * D_MODEL;
    #pragma unroll
    for (int i = 0; i < 4; ++i) {
        const int c = tid + i * 256;
        o[c] = (bf16)(v[i] * scale * w[c]);
    }
}

// ---------------------------------------------------------------------------
// BF16 WMMA GEMM:  C[M,N](f32) = A[M,K](bf16) * BT[N,K](bf16)^T  (+ epilogue)
//   EPI = 0: plain store, 1: C += Cin (residual), 2: softplus.
// Block = 256 threads = 8 waves (4 M x 2 N) of 32x64 wave tiles -> 128x128.
// K-loop staged through double-buffered LDS tiles (128x32 A + 128x32 B).
// K must be a multiple of 32.
// ---------------------------------------------------------------------------
template <int EPI>
__global__ __launch_bounds__(256)
void gemm_bf16_kernel(const bf16* __restrict__ A, const bf16* __restrict__ BT,
                      float* __restrict__ C, const float* __restrict__ Cin,
                      int K, int ldc, int ldcin) {
    __shared__ bf16 sA[2][128][LDSP];
    __shared__ bf16 sB[2][128][LDSP];

    const int tid  = threadIdx.x;
    const int lane = tid & 31;
    const int wave = tid >> 5;
    const int wm   = wave & 3;          // 0..3  (M tiles of 32)
    const int wn   = wave >> 2;         // 0..1  (N tiles of 64)
    const int lidx = lane & 15;
    const bool hi  = lane >= 16;

    // A-fragment K offsets per ISA layout (16-bit A, 16x32):
    //   lanes 0-15: K chunks {0..7, 16..23}; lanes 16-31: {8..15, 24..31}
    const int kbA = hi ? 8 : 0;
    // B-fragment: same K order as A: lanes 0-15 {0..7, 16..23}; lanes 16-31 {8..15, 24..31}.
    const int kbB = hi ? 8 : 0;

    // Cooperative staging: each thread owns one 16-element segment of one row.
    const int srow = tid & 127;
    const int scol = (tid >> 7) * 16;   // 0 or 16
    const bf16* gA = A  + (size_t)(blockIdx.x * 128 + srow) * K + scol;
    const bf16* gB = BT + (size_t)(blockIdx.y * 128 + srow) * K + scol;

    v8f acc[2][4];
    #pragma unroll
    for (int m = 0; m < 2; ++m)
        #pragma unroll
        for (int j = 0; j < 4; ++j)
            acc[m][j] = (v8f){0.f, 0.f, 0.f, 0.f, 0.f, 0.f, 0.f, 0.f};

    // Prologue: stage K-step 0 into buffer 0.
    {
        v8bf ra0 = *(const v8bf*)(gA + 0);
        v8bf ra1 = *(const v8bf*)(gA + 8);
        v8bf rb0 = *(const v8bf*)(gB + 0);
        v8bf rb1 = *(const v8bf*)(gB + 8);
        *(v8bf*)&sA[0][srow][scol]     = ra0;
        *(v8bf*)&sA[0][srow][scol + 8] = ra1;
        *(v8bf*)&sB[0][srow][scol]     = rb0;
        *(v8bf*)&sB[0][srow][scol + 8] = rb1;
    }
    __syncthreads();

    for (int k0 = 0; k0 < K; k0 += 32) {
        const int buf = (k0 >> 5) & 1;
        v8bf ra0, ra1, rb0, rb1;
        const bool more = (k0 + 32) < K;
        if (more) {
            // Issue next tile's global loads before compute (overlap).
            ra0 = *(const v8bf*)(gA + k0 + 32);
            ra1 = *(const v8bf*)(gA + k0 + 40);
            rb0 = *(const v8bf*)(gB + k0 + 32);
            rb1 = *(const v8bf*)(gB + k0 + 40);
            if (k0 + 64 < K) {
                __builtin_prefetch(gA + k0 + 64, 0, 3);
                __builtin_prefetch(gB + k0 + 64, 0, 3);
            }
        }

        // Fragments from LDS.
        const bf16* pa0 = &sA[buf][wm * 32 + lidx][kbA];
        const bf16* pa1 = &sA[buf][wm * 32 + 16 + lidx][kbA];
        const v16bf a0 = combine16(*(const v8bf*)pa0, *(const v8bf*)(pa0 + 16));
        const v16bf a1 = combine16(*(const v8bf*)pa1, *(const v8bf*)(pa1 + 16));
        #pragma unroll
        for (int j = 0; j < 4; ++j) {
            const bf16* pb = &sB[buf][wn * 64 + j * 16 + lidx][kbB];
            const v16bf b = combine16(*(const v8bf*)pb, *(const v8bf*)(pb + 16));
            acc[0][j] = __builtin_amdgcn_wmma_f32_16x16x32_bf16(
                false, a0, false, b, (short)0, acc[0][j], false, false);
            acc[1][j] = __builtin_amdgcn_wmma_f32_16x16x32_bf16(
                false, a1, false, b, (short)0, acc[1][j], false, false);
        }

        if (more) {
            const int nb = buf ^ 1;
            *(v8bf*)&sA[nb][srow][scol]     = ra0;
            *(v8bf*)&sA[nb][srow][scol + 8] = ra1;
            *(v8bf*)&sB[nb][srow][scol]     = rb0;
            *(v8bf*)&sB[nb][srow][scol + 8] = rb1;
        }
        __syncthreads();
    }

    // C/D layout: VGPR r -> M = r (lanes 0-15) or r+8 (lanes 16-31); N = lane&15.
    const int row0 = blockIdx.x * 128 + wm * 32;
    const int col0 = blockIdx.y * 128 + wn * 64;
    #pragma unroll
    for (int m = 0; m < 2; ++m) {
        const int rbase = row0 + m * 16 + (hi ? 8 : 0);
        #pragma unroll
        for (int j = 0; j < 4; ++j) {
            const int col = col0 + j * 16 + lidx;
            #pragma unroll
            for (int r = 0; r < 8; ++r) {
                const int row = rbase + r;
                float v = acc[m][j][r];
                if (EPI == 1) v += Cin[(size_t)row * ldcin + col];
                if (EPI == 2) v = softplusf(v);
                C[(size_t)row * ldc + col] = v;
            }
        }
    }
}

// ---------------------------------------------------------------------------
// Causal depthwise conv (width 4) + bias + SiLU over the first D_INNER columns
// of xr [M_ROWS, N_IN].  Writes f32 and bf16 copies of xs.
// ---------------------------------------------------------------------------
__global__ __launch_bounds__(256)
void conv_silu_kernel(const float* __restrict__ xr,
                      const float* __restrict__ conv_w,
                      const float* __restrict__ conv_b,
                      float* __restrict__ xs_f32, bf16* __restrict__ xs_bf) {
    size_t idx = (size_t)blockIdx.x * 256 + threadIdx.x;
    if (idx >= (size_t)M_ROWS * D_INNER) return;
    const int d = (int)(idx % D_INNER);
    const int l = (int)((idx / D_INNER) % SEQ);
    const int b = (int)(idx / ((size_t)D_INNER * SEQ));
    float acc = conv_b[d];
    #pragma unroll
    for (int t = 0; t < D_CONV; ++t) {
        const int ls = l + t - (D_CONV - 1);
        if (ls >= 0)
            acc += conv_w[d * D_CONV + t] *
                   xr[((size_t)(b * SEQ + ls)) * N_IN + d];
    }
    const float v = siluf(acc);
    xs_f32[idx] = v;
    xs_bf[idx]  = (bf16)v;
}

// ---------------------------------------------------------------------------
// Convert dt-input columns (0..D_INNER) of x_ssm [M_ROWS, N_X_PAD] to bf16.
// ---------------------------------------------------------------------------
__global__ __launch_bounds__(256)
void dtin_convert_kernel(const float* __restrict__ xssm, bf16* __restrict__ o) {
    size_t idx = (size_t)blockIdx.x * 256 + threadIdx.x;
    if (idx >= (size_t)M_ROWS * D_INNER) return;
    const size_t m = idx / D_INNER;
    const size_t d = idx % D_INNER;
    o[idx] = (bf16)xssm[m * N_X_PAD + d];
}

// ---------------------------------------------------------------------------
// Selective scan, software pipelined: one thread per (batch, channel d);
// 16 states in registers; next step's dt/xs preloaded to registers and B/C
// into a double-buffered LDS slot while computing the current step.
// One barrier per timestep.  grid = BATCH * (D_INNER/256) blocks of 256.
// ---------------------------------------------------------------------------
__global__ __launch_bounds__(256)
void scan_kernel(const float* __restrict__ dt, const float* __restrict__ xs,
                 const float* __restrict__ xssm, const float* __restrict__ A_log,
                 float* __restrict__ y) {
    const int tid = threadIdx.x;
    const int b   = blockIdx.x >> 3;                    // batch
    const int d   = ((blockIdx.x & 7) << 8) + tid;      // channel
    float A[D_STATE], h[D_STATE];
    #pragma unroll
    for (int n = 0; n < D_STATE; ++n) {
        A[n] = -expf(A_log[d * D_STATE + n]);
        h[n] = 0.f;
    }
    __shared__ float sB[2][D_STATE], sC[2][D_STATE];

    // Prologue: stage timestep 0 into slot 0 + registers.
    {
        const size_t r0 = (size_t)(b * SEQ) * N_X_PAD;
        if (tid < D_STATE)          sB[0][tid]           = xssm[r0 + D_INNER + tid];
        else if (tid < 2 * D_STATE) sC[0][tid - D_STATE] = xssm[r0 + D_INNER + D_STATE + (tid - D_STATE)];
    }
    float dt_c = dt[(size_t)(b * SEQ) * D_INNER + d];
    float x_c  = xs[(size_t)(b * SEQ) * D_INNER + d];
    __syncthreads();

    for (int l = 0; l < SEQ; ++l) {
        const int cur = l & 1;
        float dt_n = 0.f, x_n = 0.f;
        if (l + 1 < SEQ) {
            const size_t rn = (size_t)(b * SEQ + l + 1);
            dt_n = dt[rn * D_INNER + d];
            x_n  = xs[rn * D_INNER + d];
            if (tid < D_STATE)
                sB[cur ^ 1][tid] = xssm[rn * N_X_PAD + D_INNER + tid];
            else if (tid < 2 * D_STATE)
                sC[cur ^ 1][tid - D_STATE] = xssm[rn * N_X_PAD + D_INNER + D_STATE + (tid - D_STATE)];
        }
        float yv = 0.f;
        #pragma unroll
        for (int n = 0; n < D_STATE; ++n) {
            const float ad = expf(A[n] * dt_c);
            h[n] = ad * h[n] + dt_c * sB[cur][n] * x_c;
            yv += h[n] * sC[cur][n];
        }
        y[(size_t)(b * SEQ + l) * D_INNER + d] = yv;
        dt_c = dt_n;
        x_c  = x_n;
        __syncthreads();
    }
}

// ---------------------------------------------------------------------------
// Gating: yg = bf16( y * silu(res) ), res = xr[:, D_INNER:2*D_INNER].
// ---------------------------------------------------------------------------
__global__ __launch_bounds__(256)
void gate_kernel(const float* __restrict__ y, const float* __restrict__ xr,
                 bf16* __restrict__ yg) {
    size_t idx = (size_t)blockIdx.x * 256 + threadIdx.x;
    if (idx >= (size_t)M_ROWS * D_INNER) return;
    const size_t m = idx / D_INNER;
    const size_t d = idx % D_INNER;
    const float res = xr[m * N_IN + D_INNER + d];
    yg[idx] = (bf16)(y[idx] * siluf(res));
}

// ---------------------------------------------------------------------------
// Host-side launch
// ---------------------------------------------------------------------------
extern "C" void kernel_launch(void* const* d_in, const int* in_sizes, int n_in,
                              void* d_out, int out_size, void* d_ws, size_t ws_size,
                              hipStream_t stream) {
    (void)in_sizes; (void)n_in; (void)out_size; (void)ws_size;
    const float* x      = (const float*)d_in[0];
    const float* rms_w  = (const float*)d_in[1];
    const float* W_in   = (const float*)d_in[2];
    const float* conv_w = (const float*)d_in[3];
    const float* conv_b = (const float*)d_in[4];
    const float* W_x    = (const float*)d_in[5];
    const float* W_dt   = (const float*)d_in[6];
    const float* A_log  = (const float*)d_in[7];
    const float* W_out  = (const float*)d_in[8];
    float* out = (float*)d_out;

    char* ws = (char*)d_ws;
    size_t o = 0;
    auto take = [&](size_t bytes) { char* p = ws + o; o += (bytes + 255) & ~(size_t)255; return p; };

    bf16*  xn_bf  = (bf16*) take((size_t)M_ROWS * D_MODEL * 2);
    bf16*  WinT   = (bf16*) take((size_t)N_IN * D_MODEL * 2);
    float* xr     = (float*)take((size_t)M_ROWS * N_IN * 4);
    float* xs_f32 = (float*)take((size_t)M_ROWS * D_INNER * 4);
    bf16*  xs_bf  = (bf16*) take((size_t)M_ROWS * D_INNER * 2);
    bf16*  WxT    = (bf16*) take((size_t)N_X_PAD * D_INNER * 2);
    float* xssm   = (float*)take((size_t)M_ROWS * N_X_PAD * 4);
    bf16*  dtin   = (bf16*) take((size_t)M_ROWS * D_INNER * 2);
    bf16*  WdtT   = (bf16*) take((size_t)D_INNER * D_INNER * 2);
    float* dt     = (float*)take((size_t)M_ROWS * D_INNER * 4);
    float* ybuf   = (float*)take((size_t)M_ROWS * D_INNER * 4);
    bf16*  yg     = (bf16*) take((size_t)M_ROWS * D_INNER * 2);
    bf16*  WoutT  = (bf16*) take((size_t)D_MODEL * D_INNER * 2);

    auto blocks1d = [](size_t total) { return (unsigned)((total + 255) / 256); };

    // Weight conversions (transposed bf16), LDS-tiled transpose.
    wt_convert_kernel<<<dim3(D_MODEL / 32, N_IN / 32),    256, 0, stream>>>(W_in,  WinT,  D_MODEL, N_IN,    N_IN);
    wt_convert_kernel<<<dim3(D_INNER / 32, N_X_PAD / 32), 256, 0, stream>>>(W_x,   WxT,   D_INNER, N_X,     N_X_PAD);
    wt_convert_kernel<<<dim3(D_INNER / 32, D_INNER / 32), 256, 0, stream>>>(W_dt,  WdtT,  D_INNER, D_INNER, D_INNER);
    wt_convert_kernel<<<dim3(D_INNER / 32, D_MODEL / 32), 256, 0, stream>>>(W_out, WoutT, D_INNER, D_MODEL, D_MODEL);

    // 1. RMSNorm -> bf16
    rmsnorm_kernel<<<M_ROWS, 256, 0, stream>>>(x, rms_w, xn_bf);

    // 2. xr = xn @ W_in    [2048 x 4096]
    gemm_bf16_kernel<0><<<dim3(M_ROWS / 128, N_IN / 128), 256, 0, stream>>>(
        xn_bf, WinT, xr, nullptr, D_MODEL, N_IN, 0);

    // 3. causal depthwise conv + SiLU
    conv_silu_kernel<<<blocks1d((size_t)M_ROWS * D_INNER), 256, 0, stream>>>(
        xr, conv_w, conv_b, xs_f32, xs_bf);

    // 4. x_ssm = xs @ W_x  [2048 x 2176(pad)]
    gemm_bf16_kernel<0><<<dim3(M_ROWS / 128, N_X_PAD / 128), 256, 0, stream>>>(
        xs_bf, WxT, xssm, nullptr, D_INNER, N_X_PAD, 0);

    // 5. dt = softplus(dt_in @ W_dt)  [2048 x 2048]
    dtin_convert_kernel<<<blocks1d((size_t)M_ROWS * D_INNER), 256, 0, stream>>>(xssm, dtin);
    gemm_bf16_kernel<2><<<dim3(M_ROWS / 128, D_INNER / 128), 256, 0, stream>>>(
        dtin, WdtT, dt, nullptr, D_INNER, D_INNER, 0);

    // 6. selective scan
    scan_kernel<<<BATCH * (D_INNER / 256), 256, 0, stream>>>(dt, xs_f32, xssm, A_log, ybuf);

    // 7. gating
    gate_kernel<<<blocks1d((size_t)M_ROWS * D_INNER), 256, 0, stream>>>(ybuf, xr, yg);

    // 8. out = residual + yg @ W_out  [2048 x 1024]
    gemm_bf16_kernel<1><<<dim3(M_ROWS / 128, D_MODEL / 128), 256, 0, stream>>>(
        yg, WoutT, out, x, D_INNER, D_MODEL, D_MODEL);
}
